// TPRencoder_lstm_32796370272261
// MI455X (gfx1250) — matrix operand order repaired and checked by reading, not compiled, not run on hardware
//
#include <hip/hip_runtime.h>
#include <math.h>

// ---------- problem constants ----------
#define B_    32
#define S_    512
#define D_    768
#define H_    1024
#define KTOT  1792      // D + H (concatenated [x_t | T] GEMM)
#define NSYM  50
#define NROLE 35

// ---------- CDNA5 WMMA types ----------
typedef __attribute__((ext_vector_type(16))) __bf16        v16bf;
typedef __attribute__((ext_vector_type(8)))  float         v8f;
typedef __attribute__((ext_vector_type(4)))  unsigned int  u32x4;

// ---------- workspace layout (bytes) ----------
#define WFULL_ELEMS   (2ull * 4096ull * 1792ull)            // bf16 [cell][4H][KTOT]
#define WS_WA_OFF     (WFULL_ELEMS * 2ull)                  // 29,360,128: Wa_bf [2][64][1024] bf16
#define WS_TBF_OFF    (WS_WA_OFF + 262144ull)               // 29,622,272: T bf16 [32][1024]
#define WS_C_OFF      (WS_TBF_OFF + 65536ull)               // 29,687,808: cF then cR (f32, 32x1024 each)
#define WS_HBF_OFF    (WS_C_OFF + 262144ull)                // 29,949,952: hF then hR (bf16, 32x1024 each)

// ---------- output layout (f32 elements) ----------
#define OUT_AF_OFF 16777216ull                              // B*S*H
#define OUT_AR_OFF 17596416ull                              // + B*S*NSYM

__device__ __forceinline__ unsigned short f2bf(float f) {
    unsigned int u = __float_as_uint(f);
    unsigned int r = (u + 0x7FFFu + ((u >> 16) & 1u)) >> 16;
    return (unsigned short)r;
}
__device__ __forceinline__ float sigmoidf(float x) { return 1.0f / (1.0f + __expf(-x)); }

// =====================================================================
// One-time: fuse [Wih | Whh] per cell into a bf16 [4096][1792] matrix.
// =====================================================================
__global__ __launch_bounds__(256) void convert_weights(
    const float* __restrict__ WihF, const float* __restrict__ WhhF,
    const float* __restrict__ WihR, const float* __restrict__ WhhR,
    unsigned short* __restrict__ Wfull)
{
    unsigned long long base = (unsigned long long)blockIdx.x * (256ull * 16ull) + threadIdx.x;
    for (int i = 0; i < 16; ++i) {
        unsigned long long e = base + (unsigned long long)i * 256ull;
        if (e < WFULL_ELEMS) {
            unsigned int cell = (unsigned int)(e / (4096ull * 1792ull));
            unsigned int rem  = (unsigned int)(e % (4096ull * 1792ull));
            unsigned int n = rem / 1792u, k = rem % 1792u;
            const float* Wih = cell ? WihR : WihF;
            const float* Whh = cell ? WhhR : WhhF;
            float v = (k < 768u) ? Wih[n * 768u + k] : Whh[n * 1024u + (k - 768u)];
            Wfull[e] = f2bf(v);
        }
    }
}

// One-time: attention-head weights -> bf16, zero-padded to [2][64][1024].
__global__ __launch_bounds__(256) void convert_heads(
    const float* __restrict__ WaF_w, const float* __restrict__ WaR_w,
    unsigned short* __restrict__ Wa_bf)
{
    unsigned int base = blockIdx.x * (256u * 4u) + threadIdx.x;
    for (int i = 0; i < 4; ++i) {
        unsigned int e = base + i * 256u;
        if (e < 2u * 64u * 1024u) {
            unsigned int head = e >> 16;            // 0 = F, 1 = R
            unsigned int n    = (e >> 10) & 63u;
            unsigned int k    = e & 1023u;
            float v = 0.0f;
            if (head == 0) { if (n < NSYM)  v = WaF_w[n * 1024u + k]; }
            else           { if (n < NROLE) v = WaR_w[n * 1024u + k]; }
            Wa_bf[e] = f2bf(v);
        }
    }
}

// Zero T_bf (64KB) + cF + cR (256KB) = 81920 dwords, contiguous in ws.
__global__ __launch_bounds__(256) void init_state(unsigned int* __restrict__ p)
{
    unsigned int i = blockIdx.x * 256u + threadIdx.x;
    if (i < 81920u) p[i] = 0u;
}

// =====================================================================
// Per-step kernel 1: gates = [x_t | T] @ Wfull.T  (+bias), LSTM update.
// Grid: 64 blocks = 2 cells x 32 H-slices (32 cols each). 256 threads.
// Wave w owns gate-column tile w and both 16-row tiles -> 2 x
// v_wmma_f32_16x16x32_bf16 per K-chunk, 112 WMMA per wave per step.
// Weights stream from L2 (bf16, 29.4MB resident); activations staged
// in LDS in 256-wide K superchunks. h is emitted in bf16 (A-ready).
// =====================================================================
__global__ __launch_bounds__(256) void lstm_gates_step(
    const float*          __restrict__ x,
    const unsigned short* __restrict__ Wfull,
    const unsigned short* __restrict__ Tbf,
    float*                __restrict__ c_ws,
    unsigned short*       __restrict__ h_bf,
    const float* __restrict__ bihF, const float* __restrict__ bhhF,
    const float* __restrict__ bihR, const float* __restrict__ bhhR,
    int s)
{
    __shared__ unsigned short a_lds[32 * 256];   // 16 KB: A superchunk, bf16
    __shared__ float          g_lds[32 * 128];   // 16 KB: gate tile exchange

    const int tid  = threadIdx.x;
    const int wave = tid >> 5;
    const int lane = tid & 31;
    const int ln15 = lane & 15;
    const int hi   = lane >> 4;                  // 0: lanes 0-15, 1: lanes 16-31

    const int cell = blockIdx.x >> 5;
    const int n0   = (blockIdx.x & 31) << 5;     // H-slice start (32 wide)

    const unsigned short* Wc  = Wfull + (unsigned long long)cell * 4096ull * 1792ull;
    const float* bih = cell ? bihR : bihF;
    const float* bhh = cell ? bhhR : bhhF;
    float*          cw = c_ws + cell * (32 * 1024);
    unsigned short* hb = h_bf + cell * (32 * 1024);

    const int q   = wave >> 1;                   // gate 0..3 (i,f,g,o)
    const int off = (wave & 1) << 4;             // 0 or 16 within slice
    const int ng  = q * 1024 + n0 + off + ln15;  // row of Wfull this lane feeds (B: N=lane%16)

    v8f acc0 = {};   // rows M = 0..15
    v8f acc1 = {};   // rows M = 16..31

    for (int sc = 0; sc < 7; ++sc) {             // 7 superchunks x 256 = KTOT
        const int kbase = sc << 8;
        __syncthreads();
        // Stage A superchunk: rows = batch (32), cols = 256 K values (bf16).
        // sc 0..2 -> x_t (f32 -> bf16 on the fly); sc 3..6 -> T (already bf16).
        for (int e = tid; e < 32 * 256; e += 256) {
            const int row = e >> 8;
            const int kl  = e & 255;
            const int k   = kbase + kl;
            unsigned short v;
            if (sc < 3) v = f2bf(x[((unsigned long long)row * S_ + s) * D_ + k]);
            else        v = Tbf[row * 1024 + (k - 768)];
            a_lds[row * 256 + kl] = v;
        }
        __syncthreads();

        for (int c8 = 0; c8 < 8; ++c8) {         // 8 x K=32 chunks per superchunk
            const int k0l = c8 << 5;
            const int kg  = kbase + k0l;

            // B fragment from weight row `ng`, same K order as A:
            // (elems 0-7 K=kg+(hi?8:0)..+7, elems 8-15 +16 more).
            union { v16bf v; u32x4 q2[2]; } bfr;
            const unsigned short* bp = Wc + (unsigned long long)ng * 1792ull + kg + (hi << 3);
            bfr.q2[0] = *(const u32x4*)(bp);
            bfr.q2[1] = *(const u32x4*)(bp + 16);
            __builtin_prefetch(bp + 32, 0, 1);   // next K chunk of this weight row

            // A fragments from LDS per 16-bit A layout:
            // VGPR0-3: K = k0l + (hi?8:0) .. +7 ; VGPR4-7: +16 more.
            union { v16bf v; u32x4 q2[2]; } a0, a1;
            const unsigned short* ap0 = a_lds + ln15 * 256 + k0l + (hi << 3);
            a0.q2[0] = *(const u32x4*)(ap0);
            a0.q2[1] = *(const u32x4*)(ap0 + 16);
            const unsigned short* ap1 = ap0 + 16 * 256;    // rows 16..31
            a1.q2[0] = *(const u32x4*)(ap1);
            a1.q2[1] = *(const u32x4*)(ap1 + 16);

            acc0 = __builtin_amdgcn_wmma_f32_16x16x32_bf16(
                       false, a0.v, false, bfr.v, (short)0, acc0, false, false);
            acc1 = __builtin_amdgcn_wmma_f32_16x16x32_bf16(
                       false, a1.v, false, bfr.v, (short)0, acc1, false, false);
        }
    }

    __syncthreads();
    // C/D layout: VGPR r -> M=r (lanes 0-15) / M=r+8 (lanes 16-31), N=lane%16.
    const int col = q * 32 + off + ln15;
#pragma unroll
    for (int r = 0; r < 8; ++r) {
        const int mr = hi ? (r + 8) : r;
        g_lds[mr * 128 + col]        = acc0[r];
        g_lds[(mr + 16) * 128 + col] = acc1[r];
    }
    __syncthreads();

    // Elementwise LSTM over the 32x32 slice; c persists in workspace (f32),
    // h is emitted bf16, already row-major A-layout for the head GEMMs.
    for (int p = tid; p < 32 * 32; p += 256) {
        const int m  = p >> 5;
        const int hc = p & 31;
        const int gb = n0 + hc;
        float iv = g_lds[m * 128 +   0 + hc] + bih[        gb] + bhh[        gb];
        float fv = g_lds[m * 128 +  32 + hc] + bih[1024 + gb] + bhh[1024 + gb];
        float gv = g_lds[m * 128 +  64 + hc] + bih[2048 + gb] + bhh[2048 + gb];
        float ov = g_lds[m * 128 +  96 + hc] + bih[3072 + gb] + bhh[3072 + gb];
        const int idx = m * 1024 + gb;
        float cn = sigmoidf(fv) * cw[idx] + sigmoidf(iv) * tanhf(gv);
        cw[idx] = cn;
        hb[idx] = f2bf(sigmoidf(ov) * tanhf(cn));
    }
}

// =====================================================================
// Per-step kernel 2: head logits via WMMA (32x64x1024, two heads),
// softmax, item projections, outer-product binding. One block,
// 256 threads = 8 waves; wave w owns tile (mt = w&1, nt = w>>1).
// =====================================================================
__global__ __launch_bounds__(256) void tpr_bind_step(
    const unsigned short* __restrict__ h_bf,        // hF then hR (bf16)
    const unsigned short* __restrict__ Wa_bf,       // [2][64][1024] bf16, zero-padded
    const float* __restrict__ WaF_b, const float* __restrict__ WaR_b,
    const float* __restrict__ F_w,   const float* __restrict__ R_w,
    const float* __restrict__ scale_F, const float* __restrict__ scale_R,
    float* __restrict__ out, unsigned short* __restrict__ Tbf, int s)
{
    __shared__ float lF[32 * 64], lR[32 * 64];      // logits (padded cols unused)
    __shared__ float aF[32 * NSYM], aR[32 * NROLE];
    __shared__ float iF[32 * 32],   iR[32 * 32];

    const int tid  = threadIdx.x;
    const int wave = tid >> 5;
    const int lane = tid & 31;
    const int ln15 = lane & 15;
    const int hi   = lane >> 4;

    const int mt = wave & 1;                        // row tile: M = mt*16 ..
    const int nt = wave >> 1;                       // col tile: N = nt*16 ..

    const unsigned short* hF = h_bf;
    const unsigned short* hR = h_bf + 32 * 1024;
    const unsigned short* WF = Wa_bf;               // [64][1024]
    const unsigned short* WR = Wa_bf + 64 * 1024;

    // ---- F-head logits: [32x1024] @ [1024x64] -> 1 WMMA tile per wave ----
    {
        v8f acc = {};
        const unsigned short* abase = hF + (mt * 16 + ln15) * 1024 + (hi << 3);
        const unsigned short* bbase = WF + (nt * 16 + ln15) * 1024 + (hi << 3);
        for (int kk = 0; kk < 1024; kk += 32) {
            union { v16bf v; u32x4 q2[2]; } af, bf2;
            af.q2[0]  = *(const u32x4*)(abase + kk);
            af.q2[1]  = *(const u32x4*)(abase + kk + 16);
            bf2.q2[0] = *(const u32x4*)(bbase + kk);
            bf2.q2[1] = *(const u32x4*)(bbase + kk + 16);
            acc = __builtin_amdgcn_wmma_f32_16x16x32_bf16(
                      false, af.v, false, bf2.v, (short)0, acc, false, false);
        }
        const int col = nt * 16 + ln15;
#pragma unroll
        for (int r = 0; r < 8; ++r)
            lF[(mt * 16 + r + hi * 8) * 64 + col] = acc[r];
    }
    // ---- R-head logits ----
    {
        v8f acc = {};
        const unsigned short* abase = hR + (mt * 16 + ln15) * 1024 + (hi << 3);
        const unsigned short* bbase = WR + (nt * 16 + ln15) * 1024 + (hi << 3);
        for (int kk = 0; kk < 1024; kk += 32) {
            union { v16bf v; u32x4 q2[2]; } af, bf2;
            af.q2[0]  = *(const u32x4*)(abase + kk);
            af.q2[1]  = *(const u32x4*)(abase + kk + 16);
            bf2.q2[0] = *(const u32x4*)(bbase + kk);
            bf2.q2[1] = *(const u32x4*)(bbase + kk + 16);
            acc = __builtin_amdgcn_wmma_f32_16x16x32_bf16(
                      false, af.v, false, bf2.v, (short)0, acc, false, false);
        }
        const int col = nt * 16 + ln15;
#pragma unroll
        for (int r = 0; r < 8; ++r)
            lR[(mt * 16 + r + hi * 8) * 64 + col] = acc[r];
    }
    __syncthreads();

    // ---- softmax: 8 waves x 4 rows each (wave32 shfl reductions) ----
    for (int rr = 0; rr < 4; ++rr) {
        const int row = wave + rr * 8;
        {   // symbols (50)
            float v0 = (lane      < NSYM) ? lF[row * 64 + lane]      + WaF_b[lane]      : -3.0e38f;
            float v1 = (lane + 32 < NSYM) ? lF[row * 64 + lane + 32] + WaF_b[lane + 32] : -3.0e38f;
            float mx = fmaxf(v0, v1);
            for (int d = 16; d > 0; d >>= 1) mx = fmaxf(mx, __shfl_xor(mx, d, 32));
            float e0 = (lane      < NSYM) ? __expf(v0 - mx) : 0.0f;
            float e1 = (lane + 32 < NSYM) ? __expf(v1 - mx) : 0.0f;
            float ssum = e0 + e1;
            for (int d = 16; d > 0; d >>= 1) ssum += __shfl_xor(ssum, d, 32);
            const float inv = 1.0f / ssum;
            if (lane      < NSYM) aF[row * NSYM + lane]      = e0 * inv;
            if (lane + 32 < NSYM) aF[row * NSYM + lane + 32] = e1 * inv;
        }
        {   // roles (35)
            float v0 = (lane < NROLE) ? lR[row * 64 + lane] + WaR_b[lane] : -3.0e38f;
            float mx = v0;
            for (int d = 16; d > 0; d >>= 1) mx = fmaxf(mx, __shfl_xor(mx, d, 32));
            float e0 = (lane < NROLE) ? __expf(v0 - mx) : 0.0f;
            float ssum = e0;
            for (int d = 16; d > 0; d >>= 1) ssum += __shfl_xor(ssum, d, 32);
            const float inv = 1.0f / ssum;
            if (lane < NROLE) aR[row * NROLE + lane] = e0 * inv;
        }
    }
    __syncthreads();

    // ---- emit attention outputs ----
    for (int p = tid; p < 32 * NSYM; p += 256) {
        const int m = p / NSYM, j = p % NSYM;
        out[OUT_AF_OFF + ((unsigned long long)m * S_ + s) * NSYM + j] = aF[p];
    }
    for (int p = tid; p < 32 * NROLE; p += 256) {
        const int m = p / NROLE, j = p % NROLE;
        out[OUT_AR_OFF + ((unsigned long long)m * S_ + s) * NROLE + j] = aR[p];
    }

    // ---- item projections: 32x32 each, K=50 / K=35 (tiny, VALU) ----
    const float sF = scale_F[0], sR = scale_R[0];
    for (int p = tid; p < 1024; p += 256) {
        const int m = p >> 5, d = p & 31;
        float s1 = 0.0f;
        for (int j = 0; j < NSYM;  ++j) s1 += aF[m * NSYM  + j] * F_w[d * NSYM  + j];
        iF[p] = sF * s1;
        float s2 = 0.0f;
        for (int j = 0; j < NROLE; ++j) s2 += aR[m * NROLE + j] * R_w[d * NROLE + j];
        iR[p] = sR * s2;
    }
    __syncthreads();

    // ---- binding: T_next[m][sy*32+r] = itemF[m][sy] * itemR[m][r] ----
    for (int p = tid; p < 32 * 1024; p += 256) {
        const int m = p >> 10, h = p & 1023;
        const int sy = h >> 5, r = h & 31;
        const float v = iF[m * 32 + sy] * iR[m * 32 + r];
        out[((unsigned long long)m * S_ + s) * 1024ull + h] = v;
        Tbf[m * 1024 + h] = f2bf(v);
    }
}

// =====================================================================
extern "C" void kernel_launch(void* const* d_in, const int* in_sizes, int n_in,
                              void* d_out, int out_size, void* d_ws, size_t ws_size,
                              hipStream_t stream) {
    (void)in_sizes; (void)n_in; (void)out_size; (void)ws_size;
    const float* x     = (const float*)d_in[0];
    const float* WihF  = (const float*)d_in[1];
    const float* WhhF  = (const float*)d_in[2];
    const float* bihF  = (const float*)d_in[3];
    const float* bhhF  = (const float*)d_in[4];
    const float* WihR  = (const float*)d_in[5];
    const float* WhhR  = (const float*)d_in[6];
    const float* bihR  = (const float*)d_in[7];
    const float* bhhR  = (const float*)d_in[8];
    const float* WaF_w = (const float*)d_in[9];
    const float* WaF_b = (const float*)d_in[10];
    const float* WaR_w = (const float*)d_in[11];
    const float* WaR_b = (const float*)d_in[12];
    const float* F_w   = (const float*)d_in[13];
    const float* R_w   = (const float*)d_in[14];
    const float* scF   = (const float*)d_in[15];
    const float* scR   = (const float*)d_in[16];

    char* ws = (char*)d_ws;
    unsigned short* Wfull = (unsigned short*)(ws);
    unsigned short* Wa_bf = (unsigned short*)(ws + WS_WA_OFF);
    unsigned short* Tbf   = (unsigned short*)(ws + WS_TBF_OFF);
    float*          c_ws  = (float*)(ws + WS_C_OFF);
    unsigned short* h_bf  = (unsigned short*)(ws + WS_HBF_OFF);
    float*          out   = (float*)d_out;

    convert_weights<<<3584, 256, 0, stream>>>(WihF, WhhF, WihR, WhhR, Wfull);
    convert_heads<<<128, 256, 0, stream>>>(WaF_w, WaR_w, Wa_bf);
    init_state<<<320, 256, 0, stream>>>((unsigned int*)(ws + WS_TBF_OFF));

    for (int s = 0; s < S_; ++s) {
        lstm_gates_step<<<64, 256, 0, stream>>>(x, Wfull, Tbf, c_ws, h_bf,
                                                bihF, bhhF, bihR, bhhR, s);
        tpr_bind_step<<<1, 256, 0, stream>>>(h_bf, Wa_bf, WaF_b, WaR_b,
                                             F_w, R_w, scF, scR, out, Tbf, s);
    }
}
